// gmlp_mix_91027536871596
// MI455X (gfx1250) — matrix operand order repaired and checked by reading, not compiled, not run on hardware
//
#include <hip/hip_runtime.h>
#include <hip/hip_bf16.h>

// Fused gMLP-mix block for gfx1250 (CDNA5, wave32).
// One workgroup (256 threads = 8 waves) processes one 8x8 patch end-to-end in LDS.
// Channel-mixing matmuls use v_wmma_f32_16x16x32_f16 (M=out-chan, N=position, K=in-chan).

typedef __attribute__((ext_vector_type(16))) _Float16 v16h;
typedef __attribute__((ext_vector_type(8)))  float    v8f;
typedef __attribute__((ext_vector_type(2)))  _Float16 h2v;
typedef __attribute__((ext_vector_type(2)))  float    f2v;

#define LD64 72   // row stride (halves) for 64-channel ActT buffers
#define LD32 40   // row stride (halves) for 32-channel ActT buffers
#define LDK  52   // row stride (halves) for wdyn [pos][49]

__device__ __forceinline__ float gelu_f(float x) {
    return 0.5f * x * (1.0f + erff(x * 0.70710678118654752f));
}
__device__ __forceinline__ float sigm_f(float x) {
    return 1.0f / (1.0f + __expf(-x));
}

// A fragment (16x32, M x K) from global fp32 row-major weight W[ld rows of K].
// Layout per ISA 7.12.2: lane m = lane&15; lane[4] selects K+8 half; VGPR j -> K pairs.
__device__ __forceinline__ v16h load_a_gw(const float* __restrict__ W, int ld,
                                          int m0, int k0, int lane) {
    v16h a;
    const int m  = m0 + (lane & 15);
    const int kh = (lane >> 4) << 3;
#pragma unroll
    for (int j = 0; j < 8; ++j) {
        const int k = k0 + ((j >> 2) << 4) + kh + ((j & 3) << 1);
        f2v w = *(const f2v*)(W + m * ld + k);
        a[2 * j]     = (_Float16)w.x;
        a[2 * j + 1] = (_Float16)w.y;
    }
    return a;
}

// B fragment (32x16, K x N) from LDS ActT[pos][chan] f16 (row stride ld halves).
// lane n = lane&15; lane[4] selects K+8 half; VGPR j -> K pairs (same as A).
__device__ __forceinline__ v16h load_b_act(const _Float16* T, int ld,
                                           int n0, int k0, int lane) {
    v16h bf;
    const int n  = n0 + (lane & 15);
    const int kk = (lane >> 4) << 3;
#pragma unroll
    for (int j = 0; j < 8; ++j) {
        const int k = k0 + ((j >> 2) << 4) + kk + ((j & 3) << 1);
        h2v p = *(const h2v*)(T + n * ld + k);
        bf[2 * j]     = p.x;
        bf[2 * j + 1] = p.y;
    }
    return bf;
}

__device__ __forceinline__ v8f wmma_f16(v16h a, v16h b, v8f c) {
    return __builtin_amdgcn_wmma_f32_16x16x32_f16(false, a, false, b,
                                                  (short)0, c, false, false);
}

__global__ __launch_bounds__(256, 1)
void gmlp_mix_fused(const float* __restrict__ xin,
                    const float* __restrict__ w_hid,  const float* __restrict__ b_hid,
                    const float* __restrict__ xc_w1,  const float* __restrict__ xc_b1,
                    const float* __restrict__ xc_lnw, const float* __restrict__ xc_lnb,
                    const float* __restrict__ xc_dww, const float* __restrict__ xc_dwb,
                    const float* __restrict__ ci_w1,  const float* __restrict__ ci_b1,
                    const float* __restrict__ ci_lnw, const float* __restrict__ ci_lnb,
                    const float* __restrict__ ci_w2,  const float* __restrict__ ci_b2,
                    const float* __restrict__ si_w1,  const float* __restrict__ si_b1,
                    const float* __restrict__ si_lnw, const float* __restrict__ si_lnb,
                    const float* __restrict__ si_w2,  const float* __restrict__ si_b2,
                    const float* __restrict__ proj_w, const float* __restrict__ proj_b,
                    const float* __restrict__ out_w,  const float* __restrict__ out_b,
                    const float* __restrict__ sgu_lnw,const float* __restrict__ sgu_lnb,
                    const float* __restrict__ dwb_w0,
                    const float* __restrict__ idw_w1, const float* __restrict__ idw_b1,
                    const float* __restrict__ idw_bnw,const float* __restrict__ idw_bnb,
                    const float* __restrict__ idw_w2, const float* __restrict__ idw_b2,
                    const float* __restrict__ dwb_w2,
                    float* __restrict__ out) {
    // ---- LDS buffers ----
    __shared__ _Float16 sX[64 * LD64];            // y = gelu(hid) : ActT[pos][chan]
    __shared__ _Float16 sXc[64 * LD64];           // xc branch output
    __shared__ _Float16 sFin[64 * LD64];          // [xproj | x_sgu] concat
    __shared__ alignas(16) char poolA[64 * LD64 * 2]; // sInT -> sGate + sHid
    __shared__ alignas(16) char poolB[64 * LD64 * 2]; // sTmp1 -> sG2
    __shared__ _Float16 sG[64 * LD32];            // g = dwb_w0 @ gate
    __shared__ _Float16 sWdyn[64 * LDK];          // dynamic 7x7 kernels per pos
    __shared__ float sDw3[64 * 9];
    __shared__ float sPooled[64];
    __shared__ float sMisc[16];
    __shared__ float sCdw[32];
    __shared__ float sSi[64];

    _Float16* sInT  = (_Float16*)poolA;
    _Float16* sGate = (_Float16*)poolA;                  // reuse after stage 1
    float*    sHid  = (float*)(poolA + 64 * LD32 * 2);   // 2KB at offset 5120
    _Float16* sTmp1 = (_Float16*)poolB;
    _Float16* sG2   = (_Float16*)poolB;                  // reuse after dw3

    const int tid  = threadIdx.x;
    const int lane = tid & 31;
    const int wv   = tid >> 5;

    const int pid  = blockIdx.x;       // patch id, B * 32 * 32 total
    const int bb   = pid >> 10;
    const int prr  = pid & 1023;
    const int phh  = prr >> 5;
    const int pww  = prr & 31;
    const int row0 = phh << 3;
    const int col0 = pww << 3;

    if (tid < 64)  // warm WGP$/L2 for this patch's channel rows
        __builtin_prefetch(xin + (((size_t)bb * 64 + tid) * 256 + row0) * 256 + col0, 0, 1);

    // ---- Stage 0: load x patch (fp32 -> f16, transposed ActT[pos][chan]) ----
    for (int i = tid; i < 4096; i += 256) {
        const int c = i >> 6, p = i & 63;
        const int py = p >> 3, px = p & 7;
        const float v = xin[(((size_t)bb * 64 + c) * 256 + row0 + py) * 256 + col0 + px];
        sInT[p * LD64 + c] = (_Float16)v;
    }
    for (int i = tid; i < 576; i += 256) sDw3[i] = xc_dww[i];
    __syncthreads();

    // ---- Stage 1: y = gelu(w_hid @ x + b_hid)  (64x64 @ 64x64) ----
#pragma unroll
    for (int it = 0; it < 2; ++it) {
        const int t  = wv * 2 + it;
        const int m0 = (t >> 2) << 4, n0 = (t & 3) << 4;
        v8f acc = {};
#pragma unroll
        for (int kt = 0; kt < 2; ++kt) {
            v16h a  = load_a_gw(w_hid, 64, m0, kt * 32, lane);
            v16h bf = load_b_act(sInT, LD64, n0, kt * 32, lane);
            acc = wmma_f16(a, bf, acc);
        }
        const int n = n0 + (lane & 15);
#pragma unroll
        for (int r = 0; r < 8; ++r) {
            const int m = m0 + ((lane >> 4) << 3) + r;
            sX[n * LD64 + m] = (_Float16)gelu_f(acc[r] + b_hid[m]);
        }
    }
    __syncthreads();

    // ---- Stage 2a: xc1 = xc_w1 @ y + b1 ----
#pragma unroll
    for (int it = 0; it < 2; ++it) {
        const int t  = wv * 2 + it;
        const int m0 = (t >> 2) << 4, n0 = (t & 3) << 4;
        v8f acc = {};
#pragma unroll
        for (int kt = 0; kt < 2; ++kt) {
            v16h a  = load_a_gw(xc_w1, 64, m0, kt * 32, lane);
            v16h bf = load_b_act(sX, LD64, n0, kt * 32, lane);
            acc = wmma_f16(a, bf, acc);
        }
        const int n = n0 + (lane & 15);
#pragma unroll
        for (int r = 0; r < 8; ++r) {
            const int m = m0 + ((lane >> 4) << 3) + r;
            sTmp1[n * LD64 + m] = (_Float16)(acc[r] + xc_b1[m]);
        }
    }
    __syncthreads();

    // ---- Stage 2b: channel LayerNorm (64 ch) per position, in-place ----
    if (tid < 64) {
        const int p = tid;
        float s = 0.f, s2 = 0.f;
        for (int c = 0; c < 64; ++c) {
            const float v = (float)sTmp1[p * LD64 + c];
            s += v; s2 += v * v;
        }
        const float mu = s * (1.0f / 64.0f);
        const float rstd = rsqrtf(s2 * (1.0f / 64.0f) - mu * mu + 1e-5f);
        for (int c = 0; c < 64; ++c) {
            const float v = (float)sTmp1[p * LD64 + c];
            sTmp1[p * LD64 + c] = (_Float16)((v - mu) * rstd * xc_lnw[c] + xc_lnb[c]);
        }
    }
    __syncthreads();

    // ---- Stage 2c: depthwise 3x3 (zero-pad in patch) + bias + gelu -> sXc ----
    for (int i = tid; i < 4096; i += 256) {
        const int c = i >> 6, p = i & 63;
        const int py = p >> 3, px = p & 7;
        float acc = xc_dwb[c];
#pragma unroll
        for (int dy = -1; dy <= 1; ++dy) {
            const int yy = py + dy;
            if (yy < 0 || yy > 7) continue;
#pragma unroll
            for (int dx = -1; dx <= 1; ++dx) {
                const int xx = px + dx;
                if (xx < 0 || xx > 7) continue;
                acc += (float)sTmp1[(yy * 8 + xx) * LD64 + c] *
                       sDw3[c * 9 + (dy + 1) * 3 + (dx + 1)];
            }
        }
        sXc[p * LD64 + c] = (_Float16)gelu_f(acc);
    }
    __syncthreads();

    // ---- Stage 3: SE channel-interaction gate cdw[32] ----
    if (tid < 64) {
        float s = 0.f;
        for (int p = 0; p < 64; ++p) s += (float)sXc[p * LD64 + tid];
        sPooled[tid] = s * (1.0f / 64.0f);
    }
    __syncthreads();
    if (tid < 8) {
        float h = ci_b1[tid];
        for (int c = 0; c < 64; ++c) h += ci_w1[tid * 64 + c] * sPooled[c];
        sMisc[tid] = h;
    }
    __syncthreads();
    if (tid < 32) {
        float s = 0.f, s2 = 0.f;
        for (int q = 0; q < 8; ++q) { const float v = sMisc[q]; s += v; s2 += v * v; }
        const float mu = s * 0.125f;
        const float rstd = rsqrtf(s2 * 0.125f - mu * mu + 1e-5f);
        float acc = ci_b2[tid];
        for (int q = 0; q < 8; ++q) {
            const float hq = gelu_f((sMisc[q] - mu) * rstd * ci_lnw[q] + ci_lnb[q]);
            acc += ci_w2[tid * 8 + q] * hq;
        }
        sCdw[tid] = sigm_f(acc);
    }
    __syncthreads();

    // ---- Stage 4a: gate = LN_32(y[:,32:] * cdw) ----
    if (tid < 64) {
        const int p = tid;
        float s = 0.f, s2 = 0.f;
        for (int c = 0; c < 32; ++c) {
            const float g = (float)sX[p * LD64 + 32 + c] * sCdw[c];
            s += g; s2 += g * g;
        }
        const float mu = s * (1.0f / 32.0f);
        const float rstd = rsqrtf(s2 * (1.0f / 32.0f) - mu * mu + 1e-5f);
        for (int c = 0; c < 32; ++c) {
            const float g = (float)sX[p * LD64 + 32 + c] * sCdw[c];
            sGate[p * LD32 + c] = (_Float16)((g - mu) * rstd * sgu_lnw[c] + sgu_lnb[c]);
        }
    }
    __syncthreads();

    // ---- Stage 4b: g = dwb_w0 @ gate  (32x32 @ 32x64), one tile per wave ----
    {
        const int m0 = (wv >> 2) << 4, n0 = (wv & 3) << 4;
        v8f acc = {};
        v16h a  = load_a_gw(dwb_w0, 32, m0, 0, lane);
        v16h bf = load_b_act(sGate, LD32, n0, 0, lane);
        acc = wmma_f16(a, bf, acc);
        const int n = n0 + (lane & 15);
#pragma unroll
        for (int r = 0; r < 8; ++r) {
            const int m = m0 + ((lane >> 4) << 3) + r;
            sG[n * LD32 + m] = (_Float16)acc[r];
        }
    }
    __syncthreads();

    // ---- Stage 5a: hidden = relu(BN(idw_w1 @ g + b1)) ----
    {
        const float bnscale = rsqrtf(1.0f + 1e-5f);
        for (int i = tid; i < 512; i += 256) {
            const int q = i & 7, n = i >> 3;
            float h = idw_b1[q];
            for (int c = 0; c < 32; ++c) h += idw_w1[q * 32 + c] * (float)sG[n * LD32 + c];
            h = h * (bnscale * idw_bnw[q]) + idw_bnb[q];
            sHid[n * 8 + q] = h > 0.f ? h : 0.f;
        }
    }
    __syncthreads();

    // ---- Stage 5b: wdyn = idw_w2 @ hidden + b2  (49 per position) ----
    for (int i = tid; i < 3136; i += 256) {
        const int k = i % 49, n = i / 49;
        float v = idw_b2[k];
#pragma unroll
        for (int q = 0; q < 8; ++q) v += idw_w2[k * 8 + q] * sHid[n * 8 + q];
        sWdyn[n * LDK + k] = (_Float16)v;
    }
    __syncthreads();

    // ---- Stage 6: content-adaptive depthwise 7x7 (shared across channels) ----
    for (int i = tid; i < 2048; i += 256) {
        const int c = i & 31, n = i >> 5;
        const int py = n >> 3, px = n & 7;
        float acc = 0.f;
#pragma unroll
        for (int ky = 0; ky < 7; ++ky) {
            const int yy = py + ky - 3;
            if (yy < 0 || yy > 7) continue;
#pragma unroll
            for (int kx = 0; kx < 7; ++kx) {
                const int xx = px + kx - 3;
                if (xx < 0 || xx > 7) continue;
                acc += (float)sG[(yy * 8 + xx) * LD32 + c] *
                       (float)sWdyn[n * LDK + ky * 7 + kx];
            }
        }
        sG2[n * LD32 + c] = (_Float16)acc;
    }
    __syncthreads();

    // ---- Stage 7: x_sgu = (dwb_w2 @ gconv) * res  -> sFin[:,32:64] ----
    {
        const int m0 = (wv >> 2) << 4, n0 = (wv & 3) << 4;
        v8f acc = {};
        v16h a  = load_a_gw(dwb_w2, 32, m0, 0, lane);
        v16h bf = load_b_act(sG2, LD32, n0, 0, lane);
        acc = wmma_f16(a, bf, acc);
        const int n = n0 + (lane & 15);
#pragma unroll
        for (int r = 0; r < 8; ++r) {
            const int m = m0 + ((lane >> 4) << 3) + r;
            const float v = acc[r] * (float)sX[n * LD64 + m];  // res = y[:, :32]
            sFin[n * LD64 + 32 + m] = (_Float16)v;
        }
    }
    __syncthreads();

    // ---- Stage 8: spatial-interaction gate si[pos] ----
    if (tid < 64) {
        const int p = tid;
        float h4[4];
#pragma unroll
        for (int q = 0; q < 4; ++q) {
            float a = si_b1[q];
            for (int c = 0; c < 32; ++c)
                a += si_w1[q * 32 + c] * (float)sFin[p * LD64 + 32 + c];
            h4[q] = a;
        }
        float s = 0.f, s2 = 0.f;
#pragma unroll
        for (int q = 0; q < 4; ++q) { s += h4[q]; s2 += h4[q] * h4[q]; }
        const float mu = s * 0.25f;
        const float rstd = rsqrtf(s2 * 0.25f - mu * mu + 1e-5f);
        float acc = si_b2[0];
#pragma unroll
        for (int q = 0; q < 4; ++q)
            acc += si_w2[q] * gelu_f((h4[q] - mu) * rstd * si_lnw[q] + si_lnb[q]);
        sSi[p] = sigm_f(acc);
    }
    __syncthreads();

    // ---- Stage 9: xproj = (proj_w @ xc + b) * si  -> sFin[:,0:32] ----
    {
        const int m0 = (wv >> 2) << 4, n0 = (wv & 3) << 4;
        v8f acc = {};
#pragma unroll
        for (int kt = 0; kt < 2; ++kt) {
            v16h a  = load_a_gw(proj_w, 64, m0, kt * 32, lane);
            v16h bf = load_b_act(sXc, LD64, n0, kt * 32, lane);
            acc = wmma_f16(a, bf, acc);
        }
        const int n = n0 + (lane & 15);
#pragma unroll
        for (int r = 0; r < 8; ++r) {
            const int m = m0 + ((lane >> 4) << 3) + r;
            sFin[n * LD64 + m] = (_Float16)((acc[r] + proj_b[m]) * sSi[n]);
        }
    }
    __syncthreads();

    // ---- Stage 10: out = out_w @ [xproj | x_sgu] + out_b -> global NCHW ----
#pragma unroll
    for (int it = 0; it < 2; ++it) {
        const int t  = wv * 2 + it;
        const int m0 = (t >> 2) << 4, n0 = (t & 3) << 4;
        v8f acc = {};
#pragma unroll
        for (int kt = 0; kt < 2; ++kt) {
            v16h a  = load_a_gw(out_w, 64, m0, kt * 32, lane);
            v16h bf = load_b_act(sFin, LD64, n0, kt * 32, lane);
            acc = wmma_f16(a, bf, acc);
        }
        const int n  = n0 + (lane & 15);
        const int py = n >> 3, px = n & 7;
#pragma unroll
        for (int r = 0; r < 8; ++r) {
            const int m = m0 + ((lane >> 4) << 3) + r;
            out[(((size_t)bb * 64 + m) * 256 + row0 + py) * 256 + col0 + px] =
                acc[r] + out_b[m];
        }
    }
}

extern "C" void kernel_launch(void* const* d_in, const int* in_sizes, int n_in,
                              void* d_out, int out_size, void* d_ws, size_t ws_size,
                              hipStream_t stream) {
    (void)n_in; (void)out_size; (void)d_ws; (void)ws_size;
    const float* xin     = (const float*)d_in[0];
    const float* w_hid   = (const float*)d_in[1];
    const float* b_hid   = (const float*)d_in[2];
    const float* xc_w1   = (const float*)d_in[3];
    const float* xc_b1   = (const float*)d_in[4];
    const float* xc_lnw  = (const float*)d_in[5];
    const float* xc_lnb  = (const float*)d_in[6];
    const float* xc_dww  = (const float*)d_in[7];
    const float* xc_dwb  = (const float*)d_in[8];
    const float* ci_w1   = (const float*)d_in[9];
    const float* ci_b1   = (const float*)d_in[10];
    const float* ci_lnw  = (const float*)d_in[11];
    const float* ci_lnb  = (const float*)d_in[12];
    const float* ci_w2   = (const float*)d_in[13];
    const float* ci_b2   = (const float*)d_in[14];
    const float* si_w1   = (const float*)d_in[15];
    const float* si_b1   = (const float*)d_in[16];
    const float* si_lnw  = (const float*)d_in[17];
    const float* si_lnb  = (const float*)d_in[18];
    const float* si_w2   = (const float*)d_in[19];
    const float* si_b2   = (const float*)d_in[20];
    const float* proj_w  = (const float*)d_in[21];
    const float* proj_b  = (const float*)d_in[22];
    const float* out_w   = (const float*)d_in[23];
    const float* out_b   = (const float*)d_in[24];
    const float* sgu_lnw = (const float*)d_in[25];
    const float* sgu_lnb = (const float*)d_in[26];
    const float* dwb_w0  = (const float*)d_in[27];
    const float* idw_w1  = (const float*)d_in[28];
    const float* idw_b1  = (const float*)d_in[29];
    const float* idw_bnw = (const float*)d_in[30];
    const float* idw_bnb = (const float*)d_in[31];
    const float* idw_w2  = (const float*)d_in[32];
    const float* idw_b2  = (const float*)d_in[33];
    const float* dwb_w2  = (const float*)d_in[34];
    float* outp = (float*)d_out;

    const int nbatch  = in_sizes[0] / (64 * 256 * 256);  // = 4
    const int npatch  = nbatch * 32 * 32;                // one WG per 8x8 patch

    gmlp_mix_fused<<<npatch, 256, 0, stream>>>(
        xin, w_hid, b_hid, xc_w1, xc_b1, xc_lnw, xc_lnb, xc_dww, xc_dwb,
        ci_w1, ci_b1, ci_lnw, ci_lnb, ci_w2, ci_b2,
        si_w1, si_b1, si_lnw, si_lnb, si_w2, si_b2,
        proj_w, proj_b, out_w, out_b, sgu_lnw, sgu_lnb,
        dwb_w0, idw_w1, idw_b1, idw_bnw, idw_bnb, idw_w2, idw_b2, dwb_w2,
        outp);
}
